// KANEmbeddings_63917703299306
// MI455X (gfx1250) — matrix operand order repaired and checked by reading, not compiled, not run on hardware
//
#include <hip/hip_runtime.h>

#define B_DIM 2048
#define NF    256
#define G     64
#define D     512
#define BM    64     // b rows per workgroup
#define BN    128    // d cols per workgroup
#define LDSB_STRIDE 72   // 64 K-values + 8 pad (f16 elems) per column row

typedef _Float16 v16h __attribute__((ext_vector_type(16)));
typedef _Float16 h8   __attribute__((ext_vector_type(8)));
typedef float    v8f  __attribute__((ext_vector_type(8)));

union FragH {
    v16h v;
    h8   h[2];
    _Float16 e[16];
};

// out[b, i, d] = silu(x[b,i]) * base_w[i,d] + sum_g exp(-0.5*(x[b,i]-grid[g])^2) * gp_w[i,g,d]
__global__ __launch_bounds__(256)
void kan_embed_kernel(const float* __restrict__ x,
                      const float* __restrict__ base_w,
                      const float* __restrict__ gp_w,
                      const float* __restrict__ grid,
                      float* __restrict__ out)
{
    __shared__ _Float16 lds_b[BN * LDSB_STRIDE];  // gp_w[i] tile, transposed [n][g], f16
    __shared__ float lds_x[BM];
    __shared__ float lds_silu[BM];
    __shared__ float lds_grid[G];
    __shared__ float lds_bw[BN];

    const int t  = threadIdx.x;
    const int i  = blockIdx.z;
    const int b0 = blockIdx.x * BM;
    const int d0 = blockIdx.y * BN;

    // ---- stage x slice, silu(x), grid ----
    if (t < BM) {
        float xv = __builtin_nontemporal_load(&x[(size_t)(b0 + t) * NF + i]);
        lds_x[t] = xv;
        float en = __builtin_amdgcn_exp2f(-1.44269504f * xv);   // e^{-x}
        lds_silu[t] = xv / (1.0f + en);
        lds_grid[t] = grid[t];                                  // BM == G == 64
    }
    if (t < BN) {
        lds_bw[t] = base_w[(size_t)i * D + d0 + t];
    }

    // ---- stage gp_w[i][:, d0:d0+BN] -> LDS transposed [n][g] as f16 ----
    // 64 x 128 f32 = 2048 float4, 8 per thread, coalesced global reads (L2-resident).
    #pragma unroll
    for (int r = 0; r < 8; ++r) {
        int idx4 = t + 256 * r;           // 32 float4 per g-row
        int g    = idx4 >> 5;
        int nq   = idx4 & 31;
        const float4 v = *(const float4*)(gp_w + ((size_t)i * G + g) * D + d0 + nq * 4);
        int n = nq * 4;
        lds_b[(n + 0) * LDSB_STRIDE + g] = (_Float16)v.x;
        lds_b[(n + 1) * LDSB_STRIDE + g] = (_Float16)v.y;
        lds_b[(n + 2) * LDSB_STRIDE + g] = (_Float16)v.z;
        lds_b[(n + 3) * LDSB_STRIDE + g] = (_Float16)v.w;
    }
    __syncthreads();

    const int lane = t & 31;
    const int w    = t >> 5;
    const int tm   = w >> 1;          // 0..3 : 16-row tile
    const int tnb  = (w & 1) * 4;     // 0 or 4 : first of four 16-col tiles
    const int l15  = lane & 15;
    const int hk   = lane >> 4;

    // ---- A fragments (gp_feat) built in registers per ISA 16-bit A 16x32 layout:
    // lanes 0-15: M=lane, e0..7 -> K+0..7,  e8..15 -> K+16..23
    // lanes16-31: M=lane-16, K chunks shifted by +8
    const int   mloc  = tm * 16 + l15;
    const float xv    = lds_x[mloc];
    const int   kbase = hk * 8;
    FragH a0, a1;
    #pragma unroll
    for (int e = 0; e < 16; ++e) {
        int kk   = kbase + ((e < 8) ? e : (e + 8));
        float dA = xv - lds_grid[kk];        // K block [0,32)
        float dB = xv - lds_grid[kk + 32];   // K block [32,64)
        a0.e[e] = (_Float16)__builtin_amdgcn_exp2f(-0.72134752f * dA * dA); // exp(-0.5 d^2)
        a1.e[e] = (_Float16)__builtin_amdgcn_exp2f(-0.72134752f * dB * dB);
    }

    // ---- 4 tiles per wave: two chained WMMAs each (K=64) ----
    v8f acc[4];
    #pragma unroll
    for (int tt = 0; tt < 4; ++tt) {
        int nloc = (tnb + tt) * 16 + l15;
        // B layout: lane = N, e0..7 -> K hk*8+0..7, e8..15 -> K 16+hk*8+0..7 (matches A)
        const h8* bp = (const h8*)&lds_b[nloc * LDSB_STRIDE + hk * 8];
        FragH bf0, bf1;
        bf0.h[0] = bp[0]; bf0.h[1] = bp[2];   // K 0..31
        bf1.h[0] = bp[4]; bf1.h[1] = bp[6];   // K 32..63 (+32 halves = +4 h8)
        v8f c = {};
        c = __builtin_amdgcn_wmma_f32_16x16x32_f16(false, a0.v, false, bf0.v,
                                                   (short)0, c, false, false);
        c = __builtin_amdgcn_wmma_f32_16x16x32_f16(false, a1.v, false, bf1.v,
                                                   (short)0, c, false, false);
        acc[tt] = c;
    }

    // ---- epilogue: + silu(x)*base_w; non-temporal streaming stores (1 GiB total,
    // never re-read -> keep the gp_weight working set resident in L2)
    // C layout: VGPR j -> M = tm*16 + hk*8 + j ; N = l15
    const int mrow = tm * 16 + hk * 8;
    #pragma unroll
    for (int tt = 0; tt < 4; ++tt) {
        int   nloc = (tnb + tt) * 16 + l15;
        float bwv  = lds_bw[nloc];
        size_t base = ((size_t)(b0 + mrow) * NF + i) * D + (size_t)(d0 + nloc);
        #pragma unroll
        for (int j = 0; j < 8; ++j) {
            float sv = lds_silu[mrow + j];
            __builtin_nontemporal_store(acc[tt][j] + sv * bwv,
                                        &out[base + (size_t)j * ((size_t)NF * D)]);
        }
    }
}

extern "C" void kernel_launch(void* const* d_in, const int* in_sizes, int n_in,
                              void* d_out, int out_size, void* d_ws, size_t ws_size,
                              hipStream_t stream) {
    const float* x      = (const float*)d_in[0];
    const float* base_w = (const float*)d_in[1];
    const float* gp_w   = (const float*)d_in[2];
    const float* grid   = (const float*)d_in[3];
    float* out = (float*)d_out;

    dim3 grd(B_DIM / BM, D / BN, NF);   // (32, 4, 256) — i slowest for L2 reuse of gp_w[i]
    kan_embed_kernel<<<grd, 256, 0, stream>>>(x, base_w, gp_w, grid, out);
}
